// GATv2Model_10969346474386
// MI455X (gfx1250) — compile-verified, matrix operand order checked by reading, not run on hardware
//
#include <hip/hip_runtime.h>

typedef __attribute__((ext_vector_type(16))) _Float16 v16h;
typedef __attribute__((ext_vector_type(8)))  float    v8f;

#define NEG_BIG (-3.0e38f)

// ---------------------------------------------------------------------------
// helpers
// ---------------------------------------------------------------------------
__device__ __forceinline__ void atomicMaxF(float* addr, float val) {
  // orderable-int trick: works for mixed signs with init = -BIG
  if (val >= 0.0f) atomicMax((int*)addr, __float_as_int(val));
  else             atomicMin((unsigned int*)addr, __float_as_uint(val));
}

__global__ void fill_kernel(float* __restrict__ p, int n, float v) {
  int i = blockIdx.x * blockDim.x + threadIdx.x;
  if (i < n) p[i] = v;
}

// ---------------------------------------------------------------------------
// Pack a [K x NT*16] f32 weight matrix into per-lane WMMA B-fragment order:
// fragment (ks, t) for lane l is 16 contiguous halves at P[((ks*NT+t)*32+l)*16].
// B 32x16 f16 layout (ISA 7.12.2): lane holds col l&15, K-offsets (l>>4)*8.
// ---------------------------------------------------------------------------
__global__ void pack_w_kernel(const float* __restrict__ W, _Float16* __restrict__ P,
                              int K, int NT) {
  const int NO = NT * 16;
  int idx = blockIdx.x * blockDim.x + threadIdx.x;    // one thread per (ks,t,lane)
  int total = (K / 32) * NT * 32;
  if (idx >= total) return;
  int lane = idx & 31;
  int t    = (idx >> 5) % NT;
  int ks   = (idx >> 5) / NT;
  int lc = lane & 15;
  int ko = (lane >> 4) * 8;
  int c  = t * 16 + lc;
  int kk = ks * 32;
  _Float16* o = P + (size_t)idx * 16;
#pragma unroll
  for (int i = 0; i < 8; ++i) {
    o[i]     = (_Float16)W[(size_t)(kk + ko + i)      * NO + c];
    o[8 + i] = (_Float16)W[(size_t)(kk + 16 + ko + i) * NO + c];
  }
}

// ---------------------------------------------------------------------------
// Dual WMMA GEMM:  Yl = X@Wl + bl ; Yr = X@Wr + br
// X: [M x K] f32; BlP/BrP: pre-packed f16 B-fragments; Y: [M x NT*16] f32.
// One wave per 16-row tile; A fragment shared between the two weight matrices.
// ---------------------------------------------------------------------------
template<int K, int NT>
__global__ void dual_gemm_wmma(const float* __restrict__ X,
                               const _Float16* __restrict__ BlP, const float* __restrict__ bl,
                               const _Float16* __restrict__ BrP, const float* __restrict__ br,
                               float* __restrict__ Yl, float* __restrict__ Yr, int M) {
  const int NO = NT * 16;
  const int wave = threadIdx.x >> 5;
  const int lane = threadIdx.x & 31;
  const int rowBase = (blockIdx.x * (blockDim.x >> 5) + wave) * 16;
  if (rowBase >= M) return;

  const int lc = lane & 15;        // A-row / C-col within tile
  const int lh = lane >> 4;        // half-wave selector
  const int ko = lh * 8;           // K sub-offset inside each 16-wide K half

  v8f accL[NT] = {};
  v8f accR[NT] = {};

  int arow = rowBase + lc;
  if (arow >= M) arow = M - 1;     // clamp; padded rows discarded at store
  const float* xp = X + (size_t)arow * K;
  const v16h* bl16 = (const v16h*)BlP;
  const v16h* br16 = (const v16h*)BrP;

#pragma unroll
  for (int ks = 0; ks < K / 32; ++ks) {
    const int kk = ks * 32;
    // A fragment: 16x32 f16, loaded as 4x float4 (16B-aligned, contiguous)
    float4 p0 = *(const float4*)(xp + kk + ko);
    float4 p1 = *(const float4*)(xp + kk + ko + 4);
    float4 q0 = *(const float4*)(xp + kk + 16 + ko);
    float4 q1 = *(const float4*)(xp + kk + 16 + ko + 4);
    v16h a;
    a[0]  = (_Float16)p0.x; a[1]  = (_Float16)p0.y; a[2]  = (_Float16)p0.z; a[3]  = (_Float16)p0.w;
    a[4]  = (_Float16)p1.x; a[5]  = (_Float16)p1.y; a[6]  = (_Float16)p1.z; a[7]  = (_Float16)p1.w;
    a[8]  = (_Float16)q0.x; a[9]  = (_Float16)q0.y; a[10] = (_Float16)q0.z; a[11] = (_Float16)q0.w;
    a[12] = (_Float16)q1.x; a[13] = (_Float16)q1.y; a[14] = (_Float16)q1.z; a[15] = (_Float16)q1.w;
#pragma unroll
    for (int t = 0; t < NT; ++t) {
      v16h b0 = bl16[(size_t)(ks * NT + t) * 32 + lane];   // 32B coalesced fragment
      v16h b1 = br16[(size_t)(ks * NT + t) * 32 + lane];
      accL[t] = __builtin_amdgcn_wmma_f32_16x16x32_f16(
          false, a, false, b0, (short)0, accL[t], false, false);
      accR[t] = __builtin_amdgcn_wmma_f32_16x16x32_f16(
          false, a, false, b1, (short)0, accR[t], false, false);
    }
  }

  // C/D layout: VGPR v -> row rowBase + v + 8*lh, col lc (per 16-col tile)
#pragma unroll
  for (int v = 0; v < 8; ++v) {
    const int row = rowBase + v + 8 * lh;
    if (row < M) {
#pragma unroll
      for (int t = 0; t < NT; ++t) {
        const int c = t * 16 + lc;
        Yl[(size_t)row * NO + c] = accL[t][v] + bl[c];
        Yr[(size_t)row * NO + c] = accR[t][v] + br[c];
      }
    }
  }
}

// ---------------------------------------------------------------------------
// self-loop edge_attr 'mean' fill
// ---------------------------------------------------------------------------
__global__ void edge_deg_sum_kernel(const int* __restrict__ dst,
                                    const float* __restrict__ attr,
                                    float* __restrict__ deg, float* __restrict__ sat, int E) {
  int e = blockIdx.x * blockDim.x + threadIdx.x;
  if (e < E) {
    int d = dst[e];
    atomicAdd(&deg[d], 1.0f);
    atomicAdd(&sat[d], attr[e]);
  }
}

__global__ void mean_kernel(const float* __restrict__ deg, const float* __restrict__ sat,
                            float* __restrict__ mea, int n) {
  int i = blockIdx.x * blockDim.x + threadIdx.x;
  if (i < n) mea[i] = deg[i] > 0.0f ? sat[i] / fmaxf(deg[i], 1.0f) : 0.0f;
}

// ---------------------------------------------------------------------------
// layer-1 attention: 64 lanes per edge (4 heads x 16 channels), coalesced
// ---------------------------------------------------------------------------
__global__ void score1_kernel(const int* __restrict__ src, const int* __restrict__ dst,
                              const float* __restrict__ eattr, const float* __restrict__ mea,
                              const float* __restrict__ xl, const float* __restrict__ xr,
                              const float* __restrict__ We1, const float* __restrict__ att1,
                              float* __restrict__ e1, float* __restrict__ emax1,
                              int E, int Etot) {
  int t = blockIdx.x * blockDim.x + threadIdx.x;
  int edge = t >> 6;
  int c = t & 63;
  if (edge >= Etot) return;
  int s, d; float attr;
  if (edge < E) { s = src[edge]; d = dst[edge]; attr = eattr[edge]; }
  else          { s = edge - E;  d = s;         attr = mea[s]; }
  float m  = xl[(size_t)s * 64 + c] + xr[(size_t)d * 64 + c] + attr * We1[c];
  float lr = m > 0.0f ? m : 0.2f * m;
  float v  = lr * att1[c];
#pragma unroll
  for (int o = 8; o >= 1; o >>= 1) v += __shfl_xor(v, o, 16);
  if ((c & 15) == 0) {
    int h = c >> 4;
    e1[(size_t)edge * 4 + h] = v;
    atomicMaxF(&emax1[(size_t)d * 4 + h], v);
  }
}

__global__ void exp1_kernel(const int* __restrict__ dst, float* __restrict__ e1,
                            const float* __restrict__ emax1, float* __restrict__ den1,
                            int E, int Etot) {
  int idx = blockIdx.x * blockDim.x + threadIdx.x;
  if (idx >= Etot * 4) return;
  int edge = idx >> 2, h = idx & 3;
  int d = (edge < E) ? dst[edge] : (edge - E);
  float ee = __expf(e1[idx] - emax1[(size_t)d * 4 + h]);
  e1[idx] = ee;
  atomicAdd(&den1[(size_t)d * 4 + h], ee);
}

__global__ void agg1_kernel(const int* __restrict__ src, const int* __restrict__ dst,
                            const float* __restrict__ xl, const float* __restrict__ e1,
                            const float* __restrict__ den1, float* __restrict__ agg,
                            int E, int Etot) {
  int t = blockIdx.x * blockDim.x + threadIdx.x;
  int edge = t >> 6;
  int c = t & 63;
  if (edge >= Etot) return;
  int s, d;
  if (edge < E) { s = src[edge]; d = dst[edge]; } else { s = edge - E; d = s; }
  int h = c >> 4;
  float alpha = e1[(size_t)edge * 4 + h] / (den1[(size_t)d * 4 + h] + 1e-16f);
  atomicAdd(&agg[(size_t)d * 64 + c], xl[(size_t)s * 64 + c] * alpha);
}

__global__ void relu_bias_kernel(float* __restrict__ hb, const float* __restrict__ bias, int n64) {
  int i = blockIdx.x * blockDim.x + threadIdx.x;
  if (i < n64) hb[i] = fmaxf(hb[i] + bias[i & 63], 0.0f);
}

// ---------------------------------------------------------------------------
// layer-2 attention: 16 lanes per edge (1 head x 16 channels)
// ---------------------------------------------------------------------------
__global__ void score2_kernel(const int* __restrict__ src, const int* __restrict__ dst,
                              const float* __restrict__ xl2, const float* __restrict__ xr2,
                              const float* __restrict__ att2,
                              float* __restrict__ e2, float* __restrict__ emax2,
                              int E, int Etot) {
  int t = blockIdx.x * blockDim.x + threadIdx.x;
  int edge = t >> 4;
  int c = t & 15;
  if (edge >= Etot) return;
  int s, d;
  if (edge < E) { s = src[edge]; d = dst[edge]; } else { s = edge - E; d = s; }
  float m  = xl2[(size_t)s * 16 + c] + xr2[(size_t)d * 16 + c];
  float lr = m > 0.0f ? m : 0.2f * m;
  float v  = lr * att2[c];
#pragma unroll
  for (int o = 8; o >= 1; o >>= 1) v += __shfl_xor(v, o, 16);
  if (c == 0) {
    e2[edge] = v;
    atomicMaxF(&emax2[d], v);
  }
}

__global__ void exp2_kernel(const int* __restrict__ dst, float* __restrict__ e2,
                            const float* __restrict__ emax2, float* __restrict__ den2,
                            int E, int Etot) {
  int edge = blockIdx.x * blockDim.x + threadIdx.x;
  if (edge >= Etot) return;
  int d = (edge < E) ? dst[edge] : (edge - E);
  float ee = __expf(e2[edge] - emax2[d]);
  e2[edge] = ee;
  atomicAdd(&den2[d], ee);
}

__global__ void agg2_kernel(const int* __restrict__ src, const int* __restrict__ dst,
                            const float* __restrict__ xl2, const float* __restrict__ e2,
                            const float* __restrict__ den2, float* __restrict__ agg2,
                            int E, int Etot) {
  int t = blockIdx.x * blockDim.x + threadIdx.x;
  int edge = t >> 4;
  int c = t & 15;
  if (edge >= Etot) return;
  int s, d;
  if (edge < E) { s = src[edge]; d = dst[edge]; } else { s = edge - E; d = s; }
  float alpha = e2[edge] / (den2[d] + 1e-16f);
  atomicAdd(&agg2[(size_t)d * 16 + c], xl2[(size_t)s * 16 + c] * alpha);
}

__global__ void logsoftmax_kernel(const float* __restrict__ agg2, const float* __restrict__ bias2,
                                  float* __restrict__ out, int n) {
  int i = blockIdx.x * blockDim.x + threadIdx.x;
  if (i >= n) return;
  float v[16];
  float mx = NEG_BIG;
#pragma unroll
  for (int c = 0; c < 16; ++c) { v[c] = agg2[(size_t)i * 16 + c] + bias2[c]; mx = fmaxf(mx, v[c]); }
  float sum = 0.0f;
#pragma unroll
  for (int c = 0; c < 16; ++c) sum += __expf(v[c] - mx);
  float lse = mx + __logf(sum);
#pragma unroll
  for (int c = 0; c < 16; ++c) out[(size_t)i * 16 + c] = v[c] - lse;
}

// ---------------------------------------------------------------------------
// launcher
// ---------------------------------------------------------------------------
static inline int cdiv(int a, int b) { return (a + b - 1) / b; }

extern "C" void kernel_launch(void* const* d_in, const int* in_sizes, int n_in,
                              void* d_out, int out_size, void* d_ws, size_t ws_size,
                              hipStream_t stream) {
  const float* x     = (const float*)d_in[0];
  const int*   eidx  = (const int*)d_in[1];
  const float* eattr = (const float*)d_in[2];
  const float* Wl1   = (const float*)d_in[3];
  const float* bl1   = (const float*)d_in[4];
  const float* Wr1   = (const float*)d_in[5];
  const float* br1   = (const float*)d_in[6];
  const float* att1  = (const float*)d_in[7];
  const float* We1   = (const float*)d_in[8];
  const float* bias1 = (const float*)d_in[9];
  const float* Wl2   = (const float*)d_in[10];
  const float* bl2   = (const float*)d_in[11];
  const float* Wr2   = (const float*)d_in[12];
  const float* br2   = (const float*)d_in[13];
  const float* att2  = (const float*)d_in[14];
  const float* bias2 = (const float*)d_in[15];
  float* out = (float*)d_out;

  const int N    = in_sizes[0] / 128;
  const int E    = in_sizes[1] / 2;
  const int Etot = E + N;
  const int* srcA = eidx;
  const int* dstA = eidx + E;

  // workspace layout (float units; every block is a multiple of 8 floats -> 32B aligned)
  float* w = (float*)d_ws;
  float* xl1  = w; w += (size_t)N * 64;
  float* xr1  = w; w += (size_t)N * 64;
  float* hb   = w; w += (size_t)N * 64;   // agg1 target, then ReLU'd features
  float* deg  = w; w += N;
  float* sat  = w; w += N;
  float* mea  = w; w += N;
  float* e1   = w; w += (size_t)Etot * 4;
  float* emax1= w; w += (size_t)N * 4;
  float* den1 = w; w += (size_t)N * 4;
  float* xl2  = w; w += (size_t)N * 16;
  float* xr2  = w; w += (size_t)N * 16;
  float* e2   = w; w += Etot;
  float* emax2= w; w += N;
  float* den2 = w; w += N;
  float* agg2 = w; w += (size_t)N * 16;
  _Float16* wlp1 = (_Float16*)w; w += 4096;   // 128*64 halves
  _Float16* wrp1 = (_Float16*)w; w += 4096;
  _Float16* wlp2 = (_Float16*)w; w += 512;    // 64*16 halves
  _Float16* wrp2 = (_Float16*)w; w += 512;

  const int B = 256;

  // --- init + weight packing (f32 -> f16 WMMA B-fragment order) ---
  fill_kernel<<<cdiv(2 * N, B), B, 0, stream>>>(deg, 2 * N, 0.0f);      // deg+sat
  fill_kernel<<<cdiv(N * 64, B), B, 0, stream>>>(hb, N * 64, 0.0f);
  fill_kernel<<<cdiv(N * 4, B), B, 0, stream>>>(emax1, N * 4, NEG_BIG);
  fill_kernel<<<cdiv(N * 4, B), B, 0, stream>>>(den1, N * 4, 0.0f);
  fill_kernel<<<cdiv(N, B), B, 0, stream>>>(emax2, N, NEG_BIG);
  fill_kernel<<<cdiv(N, B), B, 0, stream>>>(den2, N, 0.0f);
  fill_kernel<<<cdiv(N * 16, B), B, 0, stream>>>(agg2, N * 16, 0.0f);
  pack_w_kernel<<<cdiv(512, B), B, 0, stream>>>(Wl1, wlp1, 128, 4);
  pack_w_kernel<<<cdiv(512, B), B, 0, stream>>>(Wr1, wrp1, 128, 4);
  pack_w_kernel<<<cdiv(64, B), B, 0, stream>>>(Wl2, wlp2, 64, 1);
  pack_w_kernel<<<cdiv(64, B), B, 0, stream>>>(Wr2, wrp2, 64, 1);

  // --- self-loop edge_attr mean ---
  edge_deg_sum_kernel<<<cdiv(E, B), B, 0, stream>>>(dstA, eattr, deg, sat, E);
  mean_kernel<<<cdiv(N, B), B, 0, stream>>>(deg, sat, mea, N);

  // --- layer 1: node transforms via WMMA (K=128, 4 col-tiles of 16) ---
  const int tiles = cdiv(N, 16);
  const int gemmBlocks = cdiv(tiles, B / 32);
  dual_gemm_wmma<128, 4><<<gemmBlocks, B, 0, stream>>>(x, wlp1, bl1, wrp1, br1, xl1, xr1, N);

  // --- layer 1 attention ---
  score1_kernel<<<cdiv(Etot * 64, B), B, 0, stream>>>(srcA, dstA, eattr, mea, xl1, xr1,
                                                      We1, att1, e1, emax1, E, Etot);
  exp1_kernel<<<cdiv(Etot * 4, B), B, 0, stream>>>(dstA, e1, emax1, den1, E, Etot);
  agg1_kernel<<<cdiv(Etot * 64, B), B, 0, stream>>>(srcA, dstA, xl1, e1, den1, hb, E, Etot);
  relu_bias_kernel<<<cdiv(N * 64, B), B, 0, stream>>>(hb, bias1, N * 64);

  // --- layer 2: node transforms via WMMA (K=64, 1 col-tile of 16) ---
  dual_gemm_wmma<64, 1><<<gemmBlocks, B, 0, stream>>>(hb, wlp2, bl2, wrp2, br2, xl2, xr2, N);

  // --- layer 2 attention ---
  score2_kernel<<<cdiv(Etot * 16, B), B, 0, stream>>>(srcA, dstA, xl2, xr2, att2,
                                                      e2, emax2, E, Etot);
  exp2_kernel<<<cdiv(Etot, B), B, 0, stream>>>(dstA, e2, emax2, den2, E, Etot);
  agg2_kernel<<<cdiv(Etot * 16, B), B, 0, stream>>>(srcA, dstA, xl2, e2, den2, agg2, E, Etot);

  // --- log_softmax ---
  logsoftmax_kernel<<<cdiv(N, B), B, 0, stream>>>(agg2, bias2, out, N);
}
